// FluxRoPEAttention_52956946759716
// MI455X (gfx1250) — matrix operand order repaired and checked by reading, not compiled, not run on hardware
//
#include <hip/hip_runtime.h>

typedef _Float16 v16h __attribute__((ext_vector_type(16)));
typedef _Float16 v8h  __attribute__((ext_vector_type(8)));
typedef float    v8f  __attribute__((ext_vector_type(8)));
typedef unsigned int u32x4 __attribute__((ext_vector_type(4)));
typedef int      i32x4 __attribute__((ext_vector_type(4)));
typedef int      i32x8 __attribute__((ext_vector_type(8)));

#define HD   128
#define LSEQ 4096
#define NH   24
#define BC   32            // keys per tile
#define WAVES 8
#define BR   (WAVES * 16)  // 128 query rows per block
#define NT   (LSEQ / BC)   // key tiles

// 1/sqrt(128) * log2(e): softmax runs in the log2 domain (v_exp_f32 is exp2)
#define QSCALE (0.08838834764831845f * 1.4426950408889634f)

union V16U { v16h v; v8h h[2]; };
union HPK  { _Float16 h[2]; unsigned int u; };

// ---------------- DPP16 butterfly reductions (stay within 16-lane rows) -----
#define DPPF(x, ctrl)                                                          \
    __builtin_bit_cast(float, __builtin_amdgcn_update_dpp(                     \
        0, __builtin_bit_cast(int, (x)), (ctrl), 0xf, 0xf, true))

__device__ __forceinline__ float rowmax16(float t) {
    t = fmaxf(t, DPPF(t, 0xB1));   // quad_perm [1,0,3,2]  (xor 1)
    t = fmaxf(t, DPPF(t, 0x4E));   // quad_perm [2,3,0,1]  (xor 2)
    t = fmaxf(t, DPPF(t, 0x141));  // row_half_mirror      (xor 4-equiv)
    t = fmaxf(t, DPPF(t, 0x140));  // row_mirror           (xor 8-equiv)
    return t;
}
__device__ __forceinline__ float rowsum16(float t) {
    t += DPPF(t, 0xB1);
    t += DPPF(t, 0x4E);
    t += DPPF(t, 0x141);
    t += DPPF(t, 0x140);
    return t;
}

// ---------------- TDM: 2D f16 tile DMA, global -> LDS (descriptor per ISA §8)
__device__ __forceinline__ void tdm_load_2d(unsigned int lds_off, const void* gptr,
                                            int dim0, int dim1, int tile0, int tile1,
                                            int stride0)
{
    u32x4 g0;
    g0.x = 1u;                                   // count=1, user mode
    g0.y = lds_off;                              // LDS byte address
    unsigned long long ga = (unsigned long long)(size_t)gptr;
    g0.z = (unsigned int)ga;                     // global_addr [31:0] at bits 95:64
    g0.w = (unsigned int)(ga >> 32) | (2u << 30);// global_addr [56:32] + type=2
    i32x8 g1;
    g1[0] = 0x00010000;                          // data_size=1 (2 bytes), no extras
    g1[1] = (dim0 & 0xffff) << 16;               // tensor_dim0[15:0] at bits 63:48
    g1[2] = ((dim0 >> 16) & 0xffff) | ((dim1 & 0xffff) << 16);
    g1[3] = ((dim1 >> 16) & 0xffff) | (tile0 << 16);
    g1[4] = tile1 & 0xffff;                      // tile_dim1 (tile_dim2 = 0)
    g1[5] = stride0;                             // tensor_dim0_stride[31:0]
    g1[6] = 0;
    g1[7] = 0;
    i32x4 z4 = {0, 0, 0, 0};
    i32x8 z8 = {0, 0, 0, 0, 0, 0, 0, 0};
    __builtin_amdgcn_tensor_load_to_lds(g0, g1, z4, z4, z8, 0);
}

// ---------------- shared per-tile compute: S = QK^T, online softmax, O += PV -
__device__ __forceinline__ void attn_tile(const _Float16* __restrict__ Kld,
                                          const _Float16* __restrict__ Vld,
                                          _Float16* __restrict__ Pw,
                                          const v16h (&qA)[4], v8f (&O)[8],
                                          float (&mrow)[8], float (&lrow)[8],
                                          int m16, int hi)
{
    v8f s0, s1;
    #pragma unroll
    for (int e = 0; e < 8; ++e) { s0[e] = 0.0f; s1[e] = 0.0f; }
    #pragma unroll
    for (int c = 0; c < 4; ++c) {
        // B 32x16 as loaded here (contiguous k, same as qA): lane: N=m16 (key row); element j: K = hi*16 + j
        const v16h b0 = *(const v16h*)&Kld[m16 * HD + c * 32 + hi * 16];
        s0 = __builtin_amdgcn_wmma_f32_16x16x32_f16(false, qA[c], false, b0,
                                                    (short)0, s0, false, false);
        const v16h b1 = *(const v16h*)&Kld[(16 + m16) * HD + c * 32 + hi * 16];
        s1 = __builtin_amdgcn_wmma_f32_16x16x32_f16(false, qA[c], false, b1,
                                                    (short)0, s1, false, false);
    }

    // online softmax in log2 domain (C layout: VGPR r, lane: M=r+8*hi, N=m16)
    #pragma unroll
    for (int r = 0; r < 8; ++r) {
        const float t = rowmax16(fmaxf(s0[r], s1[r]));
        const float mprev = mrow[r];
        const float mnew  = fmaxf(mprev, t);
        const float corr  = __builtin_amdgcn_exp2f(mprev - mnew);
        mrow[r] = mnew;
        const float p0 = __builtin_amdgcn_exp2f(s0[r] - mnew);
        const float p1 = __builtin_amdgcn_exp2f(s1[r] - mnew);
        const float rs = rowsum16(p0 + p1);
        lrow[r] = lrow[r] * corr + rs;
        #pragma unroll
        for (int ct = 0; ct < 8; ++ct) O[ct][r] *= corr;
        const int M = r + 8 * hi;
        Pw[M * BC + m16]      = (_Float16)p0;
        Pw[M * BC + 16 + m16] = (_Float16)p1;
    }

    // re-layout P (C -> A) through per-wave LDS scratch, then O += P * V
    V16U up;
    up.h[0] = *(const v8h*)&Pw[m16 * BC + hi * 16];
    up.h[1] = *(const v8h*)&Pw[m16 * BC + hi * 16 + 8];
    const v16h pA = up.v;
    #pragma unroll
    for (int ct = 0; ct < 8; ++ct) {
        const v16h vb = *(const v16h*)&Vld[(ct * 16 + m16) * BC + hi * 16];
        O[ct] = __builtin_amdgcn_wmma_f32_16x16x32_f16(false, pA, false, vb,
                                                       (short)0, O[ct], false, false);
    }
}

// ---------------- Q fragments: RoPE + scale(log2e) + f16, WMMA A-layout ------
__device__ __forceinline__ void load_qA(const float* __restrict__ q,
                                        const float* __restrict__ pe,
                                        int h, int qrow, int hi, v16h (&qA)[4])
{
    const float* qr = q  + ((size_t)h * LSEQ + qrow) * HD;
    const float* pr = pe + (size_t)qrow * HD;   // [L][64][2] flat == [L][128]
    #pragma unroll
    for (int c = 0; c < 4; ++c) {
        V16U u;
        #pragma unroll
        for (int g = 0; g < 2; ++g) {
            const int f0 = c * 32 + hi * 16 + g * 8;
            const float4 a0 = *(const float4*)(qr + f0);
            const float4 a1 = *(const float4*)(qr + f0 + 4);
            const float4 b0 = *(const float4*)(pr + f0);
            const float4 b1 = *(const float4*)(pr + f0 + 4);
            v8h t;
            t[0] = (_Float16)((a0.x * b0.x - a0.y * b0.y) * QSCALE);
            t[1] = (_Float16)((a0.x * b0.y + a0.y * b0.x) * QSCALE);
            t[2] = (_Float16)((a0.z * b0.z - a0.w * b0.w) * QSCALE);
            t[3] = (_Float16)((a0.z * b0.w + a0.w * b0.z) * QSCALE);
            t[4] = (_Float16)((a1.x * b1.x - a1.y * b1.y) * QSCALE);
            t[5] = (_Float16)((a1.x * b1.y + a1.y * b1.x) * QSCALE);
            t[6] = (_Float16)((a1.z * b1.z - a1.w * b1.w) * QSCALE);
            t[7] = (_Float16)((a1.z * b1.w + a1.w * b1.z) * QSCALE);
            u.h[g] = t;
        }
        qA[c] = u.v;
    }
}

// ============== Phase 1: RoPE(K)->f16 [H][L][D], V->f16 transposed [H][D][L] =
__launch_bounds__(256)
__global__ void preproc_kernel(const float* __restrict__ k,
                               const float* __restrict__ v,
                               const float* __restrict__ pe,
                               _Float16* __restrict__ kf,
                               _Float16* __restrict__ vt)
{
    __shared__ alignas(16) _Float16 Vl[HD * 64];   // [feat][key] tile, 16 KB
    const int tid = threadIdx.x;
    const int h   = blockIdx.y;
    const int l0  = blockIdx.x * 64;

    #pragma unroll
    for (int i = 0; i < 16; ++i) {                 // 64 rows * 64 pairs / 256 thr
        const int p   = i * 256 + tid;
        const int row = p >> 6;
        const int d   = p & 63;
        const size_t off = ((size_t)h * LSEQ + l0 + row) * HD + 2 * d;
        const float2 kk = *(const float2*)(k + off);
        const float2 cs = *(const float2*)(pe + (size_t)(l0 + row) * HD + 2 * d);
        HPK pk;
        pk.h[0] = (_Float16)(kk.x * cs.x - kk.y * cs.y);
        pk.h[1] = (_Float16)(kk.x * cs.y + kk.y * cs.x);
        *(unsigned int*)&kf[off] = pk.u;           // coalesced packed store
        const float2 vv = *(const float2*)(v + off);
        Vl[(2 * d) * 64 + row]     = (_Float16)vv.x;
        Vl[(2 * d + 1) * 64 + row] = (_Float16)vv.y;
    }
    __syncthreads();
    // write transposed V rows coalesced: 32 halfs (64B) per thread
    const int f  = tid >> 1;
    const int c0 = (tid & 1) * 32;
    const uint4* s = (const uint4*)&Vl[f * 64 + c0];
    uint4* dt = (uint4*)&vt[((size_t)h * HD + f) * LSEQ + l0 + c0];
    dt[0] = s[0]; dt[1] = s[1]; dt[2] = s[2]; dt[3] = s[3];
}

// ============== Phase 2: flash attention, TDM double-buffered staging ========
__launch_bounds__(WAVES * 32)
__global__ void fa_rope_tdm_kernel(const float* __restrict__ q,
                                   const _Float16* __restrict__ kf,
                                   const _Float16* __restrict__ vt,
                                   const float* __restrict__ pe,
                                   float* __restrict__ out)
{
    __shared__ alignas(32) _Float16 Kbuf[2][BC * HD];   // [key][feat]
    __shared__ alignas(32) _Float16 Vbuf[2][HD * BC];   // [feat][key]
    __shared__ alignas(32) _Float16 Pld[WAVES][16 * BC];

    const int tid  = threadIdx.x;
    const int lane = tid & 31;
    const int wave = tid >> 5;
    const int m16  = lane & 15;
    const int hi   = lane >> 4;

    const int h     = blockIdx.y;
    const int qbase = blockIdx.x * BR + wave * 16;

    v16h qA[4];
    load_qA(q, pe, h, qbase + m16, hi, qA);

    v8f O[8];
    #pragma unroll
    for (int i = 0; i < 8; ++i) {
        #pragma unroll
        for (int e = 0; e < 8; ++e) O[i][e] = 0.0f;
    }
    float mrow[8], lrow[8];
    #pragma unroll
    for (int r = 0; r < 8; ++r) { mrow[r] = -1e30f; lrow[r] = 0.0f; }

    const _Float16* kfh = kf + (size_t)h * LSEQ * HD;
    const _Float16* vth = vt + (size_t)h * HD * LSEQ;
    _Float16* Pw = &Pld[wave][0];

    const unsigned int koff0 = (unsigned int)(size_t)&Kbuf[0][0];
    const unsigned int koff1 = (unsigned int)(size_t)&Kbuf[1][0];
    const unsigned int voff0 = (unsigned int)(size_t)&Vbuf[0][0];
    const unsigned int voff1 = (unsigned int)(size_t)&Vbuf[1][0];

    if (wave == 0) {   // prologue DMA: tile 0 -> buffer 0
        tdm_load_2d(koff0, kfh, HD, BC, HD, BC, HD);
        tdm_load_2d(voff0, vth, BC, HD, BC, HD, LSEQ);
    }

    for (int t = 0; t < NT; ++t) {
        const int cur = t & 1;
        __syncthreads();   // everyone done reading buf[cur^1] (tile t-1)
        if (wave == 0) {
            if (t + 1 < NT) {   // prefetch tile t+1 into the other buffer
                tdm_load_2d(cur ? koff0 : koff1,
                            kfh + (size_t)(t + 1) * BC * HD, HD, BC, HD, BC, HD);
                tdm_load_2d(cur ? voff0 : voff1,
                            vth + (size_t)(t + 1) * BC, BC, HD, BC, HD, LSEQ);
                __builtin_amdgcn_s_wait_tensorcnt(2);  // tile t's pair complete
            } else {
                __builtin_amdgcn_s_wait_tensorcnt(0);
            }
        }
        __syncthreads();   // publish tile t to all waves

        attn_tile(&Kbuf[cur][0], &Vbuf[cur][0], Pw, qA, O, mrow, lrow, m16, hi);
    }

    #pragma unroll
    for (int r = 0; r < 8; ++r) {
        const float inv = 1.0f / lrow[r];
        float* op = out + (size_t)(qbase + r + 8 * hi) * (NH * HD) + h * HD;
        #pragma unroll
        for (int ct = 0; ct < 8; ++ct)
            op[ct * 16 + m16] = O[ct][r] * inv;
    }
}

// ============== Fallback: self-contained kernel (no workspace needed) ========
__launch_bounds__(WAVES * 32)
__global__ void fa_rope_kernel(const float* __restrict__ q,
                               const float* __restrict__ k,
                               const float* __restrict__ v,
                               const float* __restrict__ pe,
                               float* __restrict__ out)
{
    __shared__ alignas(32) _Float16 Kld[BC * HD];
    __shared__ alignas(32) _Float16 Vld[HD * BC];
    __shared__ alignas(32) _Float16 Pld[WAVES][16 * BC];

    const int tid  = threadIdx.x;
    const int lane = tid & 31;
    const int wave = tid >> 5;
    const int m16  = lane & 15;
    const int hi   = lane >> 4;

    const int h     = blockIdx.y;
    const int qbase = blockIdx.x * BR + wave * 16;

    v16h qA[4];
    load_qA(q, pe, h, qbase + m16, hi, qA);

    v8f O[8];
    #pragma unroll
    for (int i = 0; i < 8; ++i) {
        #pragma unroll
        for (int e = 0; e < 8; ++e) O[i][e] = 0.0f;
    }
    float mrow[8], lrow[8];
    #pragma unroll
    for (int r = 0; r < 8; ++r) { mrow[r] = -1e30f; lrow[r] = 0.0f; }

    const float* kh = k + (size_t)h * LSEQ * HD;
    const float* vh = v + (size_t)h * LSEQ * HD;
    _Float16* Pw = &Pld[wave][0];

    for (int kt = 0; kt < NT; ++kt) {
        __syncthreads();
        const int kbase = kt * BC;
        #pragma unroll
        for (int i = 0; i < (BC * HD / 2) / (WAVES * 32); ++i) {
            const int p   = i * (WAVES * 32) + tid;
            const int row = p >> 6;
            const int d   = p & 63;
            const size_t off = (size_t)(kbase + row) * HD + 2 * d;
            const float2 kk = *(const float2*)(kh + off);
            const float2 cs = *(const float2*)(pe + off);
            Kld[row * HD + 2 * d]     = (_Float16)(kk.x * cs.x - kk.y * cs.y);
            Kld[row * HD + 2 * d + 1] = (_Float16)(kk.x * cs.y + kk.y * cs.x);
            const float2 vv = *(const float2*)(vh + off);
            Vld[(2 * d) * BC + row]     = (_Float16)vv.x;
            Vld[(2 * d + 1) * BC + row] = (_Float16)vv.y;
        }
        __syncthreads();
        attn_tile(Kld, Vld, Pw, qA, O, mrow, lrow, m16, hi);
    }

    #pragma unroll
    for (int r = 0; r < 8; ++r) {
        const float inv = 1.0f / lrow[r];
        float* op = out + (size_t)(qbase + r + 8 * hi) * (NH * HD) + h * HD;
        #pragma unroll
        for (int ct = 0; ct < 8; ++ct)
            op[ct * 16 + m16] = O[ct][r] * inv;
    }
}

extern "C" void kernel_launch(void* const* d_in, const int* in_sizes, int n_in,
                              void* d_out, int out_size, void* d_ws, size_t ws_size,
                              hipStream_t stream) {
    (void)in_sizes; (void)n_in; (void)out_size;
    const float* q  = (const float*)d_in[0];
    const float* k  = (const float*)d_in[1];
    const float* v  = (const float*)d_in[2];
    const float* pe = (const float*)d_in[3];
    float* out = (float*)d_out;

    const size_t half_elems = (size_t)NH * LSEQ * HD;       // per tensor
    const size_t need = half_elems * 2 /*tensors*/ * 2 /*bytes*/;

    if (d_ws != nullptr && ws_size >= need) {
        _Float16* kf = (_Float16*)d_ws;
        _Float16* vt = kf + half_elems;
        preproc_kernel<<<dim3(LSEQ / 64, NH), 256, 0, stream>>>(k, v, pe, kf, vt);
        fa_rope_tdm_kernel<<<dim3(LSEQ / BR, NH), WAVES * 32, 0, stream>>>(q, kf, vt, pe, out);
    } else {
        fa_rope_kernel<<<dim3(LSEQ / BR, NH), WAVES * 32, 0, stream>>>(q, k, v, pe, out);
    }
}
